// DynamicFusion_31997506355854
// MI455X (gfx1250) — matrix operand order repaired and checked by reading, not compiled, not run on hardware
//
#include <hip/hip_runtime.h>
#include <hip/hip_bf16.h>
#include <stdint.h>

typedef __attribute__((ext_vector_type(16))) _Float16 v16h;
typedef __attribute__((ext_vector_type(8)))  float    v8f;

// L=512, B=8, N=32, D=256, GH=128, AH=64
#define OUT_PM   1048576      // fused = L*B*D floats, then per_msg
#define OUT_ATTN 34603008     // + L*B*N*D floats, then attn

union Frag { v16h h; uint32_t u[8]; };

// -----------------------------------------------------------------------------
// Prep: swizzle gw1^T (512x128) and aw1^T (256x64) into WMMA B-fragment order
// (f16). B-frag layout per 32x16 tile: lane half h holds K=8h+2j,+1 (VGPR j<4) and
// K=16+8h+2(j-4),+1 (VGPR j>=4), column n = lane&15.  ws dwords:
//   gw1: [((nt*16+kt)*32+lane)*8 + j]            nt=0..7, kt=0..15
//   aw1: 32768 + [((nt*8+kt)*32+lane)*8 + j]     nt=0..3, kt=0..7
// -----------------------------------------------------------------------------
__global__ __launch_bounds__(256) void prep_weights(const float* __restrict__ gw1,
                                                    const float* __restrict__ aw1,
                                                    uint32_t* __restrict__ ws) {
  int i = blockIdx.x * 256 + threadIdx.x;
  if (i < 32768) {
    int j = i & 7, lane = (i >> 3) & 31, kt = (i >> 8) & 15, nt = i >> 12;
    int n = nt * 16 + (lane & 15);
    int k = kt * 32 + ((lane >> 4) << 3) + 2 * (j & 3) + ((j >> 2) << 4);
    union { _Float16 h[2]; uint32_t u; } p;
    p.h[0] = (_Float16)gw1[n * 512 + k];
    p.h[1] = (_Float16)gw1[n * 512 + k + 1];
    ws[i] = p.u;
  } else if (i < 40960) {
    int t = i - 32768;
    int j = t & 7, lane = (t >> 3) & 31, kt = (t >> 8) & 7, nt = t >> 11;
    int n = nt * 16 + (lane & 15);
    int k = kt * 32 + ((lane >> 4) << 3) + 2 * (j & 3) + ((j >> 2) << 4);
    union { _Float16 h[2]; uint32_t u; } p;
    p.h[0] = (_Float16)aw1[n * 256 + k];
    p.h[1] = (_Float16)aw1[n * 256 + k + 1];
    ws[32768 + t] = p.u;
  }
}

// -----------------------------------------------------------------------------
// Main kernel: one workgroup per (l,b). 128 threads = 4 wave32.
// -----------------------------------------------------------------------------
#define CSTRIDE 520   // comb row stride (halfs): 512 + 8 pad
#define PSTRIDE 264   // per_msg row stride (halfs): 256 + 8 pad
#define HSTRIDE 136   // h row stride (halfs): 128 + 8 pad
#define ASTRIDE 72    // att row stride (halfs): 64 + 8 pad

__global__ __launch_bounds__(128) void fusion_kernel(
    const float* __restrict__ vert, const float* __restrict__ horiz,
    const unsigned char* __restrict__ mask,
    const float* __restrict__ gb1, const float* __restrict__ gw2,
    const float* __restrict__ gb2, const float* __restrict__ ab1,
    const float* __restrict__ aw2, const float* __restrict__ ab2,
    const uint32_t* __restrict__ wfrag,
    float* __restrict__ out) {
  __shared__ __align__(16) _Float16 s_comb[32 * CSTRIDE];  // 33280 B
  __shared__ __align__(16) _Float16 s_pm  [32 * PSTRIDE];  // 16896 B
  __shared__ __align__(16) _Float16 s_h   [32 * HSTRIDE];  //  8704 B
  __shared__ __align__(16) _Float16 s_att [32 * ASTRIDE];  //  4608 B
  __shared__ float s_gate[32];
  __shared__ float s_logits[32];
  __shared__ float s_attnw[32];

  const int tid  = threadIdx.x;
  const int lane = tid & 31;
  const int wave = tid >> 5;
  const int g    = blockIdx.x;   // l*B + b
  const int l    = g >> 3;
  const int b    = g & 7;

  const float4* v4 = (const float4*)vert  + (size_t)g * 2048;
  const float4* h4 = (const float4*)horiz + (size_t)g * 2048;

  // ---- Phase 1: stage concat(vert,horiz) as f16 in LDS (coalesced b128 loads)
  for (int i = tid; i < 2048; i += 128) {
    int n = i >> 6, d4 = i & 63;
    float4 vv = v4[i];
    float4 hh = h4[i];
    union { _Float16 h[4]; uint2 u; } pv, ph;
    pv.h[0] = (_Float16)vv.x; pv.h[1] = (_Float16)vv.y;
    pv.h[2] = (_Float16)vv.z; pv.h[3] = (_Float16)vv.w;
    ph.h[0] = (_Float16)hh.x; ph.h[1] = (_Float16)hh.y;
    ph.h[2] = (_Float16)hh.z; ph.h[3] = (_Float16)hh.w;
    *(uint2*)(&s_comb[n * CSTRIDE + d4 * 4])       = pv.u;
    *(uint2*)(&s_comb[n * CSTRIDE + 256 + d4 * 4]) = ph.u;
  }
  __syncthreads();

  const int s  = lane >> 4;
  const int ml = lane & 15;

  // ---- Phase 2: gate layer 1  [32,512] @ [512,128] via WMMA f16 -> relu -> s_h
  {
    v8f acc[2][2] = {};            // [mt][nti]
    const int nt0 = wave * 2;      // this wave owns columns nt0, nt0+1
    const uint32_t* r0 = (const uint32_t*)(s_comb + ml * CSTRIDE);
    const uint32_t* r1 = (const uint32_t*)(s_comb + (16 + ml) * CSTRIDE);
    for (int kt = 0; kt < 16; ++kt) {
      Frag a0, a1, b0, b1;
      int kd = (kt * 32 + s * 8) >> 1;   // dword index in row
#pragma unroll
      for (int j = 0; j < 4; ++j) {
        a0.u[j] = r0[kd + j];  a0.u[4 + j] = r0[kd + 8 + j];
        a1.u[j] = r1[kd + j];  a1.u[4 + j] = r1[kd + 8 + j];
      }
      const uint32_t* bp0 = wfrag + ((( nt0      * 16 + kt) * 32 + lane) << 3);
      const uint32_t* bp1 = wfrag + ((((nt0 + 1) * 16 + kt) * 32 + lane) << 3);
#pragma unroll
      for (int j = 0; j < 8; ++j) { b0.u[j] = bp0[j]; b1.u[j] = bp1[j]; }
      acc[0][0] = __builtin_amdgcn_wmma_f32_16x16x32_f16(false, a0.h, false, b0.h, (short)0, acc[0][0], false, false);
      acc[0][1] = __builtin_amdgcn_wmma_f32_16x16x32_f16(false, a0.h, false, b1.h, (short)0, acc[0][1], false, false);
      acc[1][0] = __builtin_amdgcn_wmma_f32_16x16x32_f16(false, a1.h, false, b0.h, (short)0, acc[1][0], false, false);
      acc[1][1] = __builtin_amdgcn_wmma_f32_16x16x32_f16(false, a1.h, false, b1.h, (short)0, acc[1][1], false, false);
    }
#pragma unroll
    for (int mt = 0; mt < 2; ++mt)
#pragma unroll
      for (int nti = 0; nti < 2; ++nti) {
        int col = (nt0 + nti) * 16 + ml;
        float bias = gb1[col];
#pragma unroll
        for (int r = 0; r < 8; ++r) {
          int m = mt * 16 + r + 8 * s;           // C layout: VGPR r, lane half s
          float hv = acc[mt][nti][r] + bias;
          s_h[m * HSTRIDE + col] = (_Float16)(hv > 0.f ? hv : 0.f);
        }
      }
  }
  __syncthreads();

  // ---- Phase 3: gate layer 2 (128-dot per token, 4 threads/token) + sigmoid
  {
    int tok = tid >> 2, q = tid & 3;
    float p = 0.f;
    const _Float16* row = s_h + tok * HSTRIDE + q * 32;
#pragma unroll
    for (int i = 0; i < 32; ++i) p += (float)row[i] * gw2[q * 32 + i];
    p += __shfl_xor(p, 1, 32);
    p += __shfl_xor(p, 2, 32);
    if (q == 0) {
      float z = p + gb2[0];
      s_gate[tok] = 1.f / (1.f + __expf(-z));
    }
  }
  __syncthreads();

  // ---- Phase 4: per_msg = g*v + (1-g)*h ; write f32 to global, f16 to LDS
  {
    float4* pm4 = (float4*)(out + OUT_PM) + (size_t)g * 2048;
    for (int i = tid; i < 2048; i += 128) {
      int n = i >> 6, d4 = i & 63;
      float gt = s_gate[n];
      float4 vv = v4[i], hh = h4[i], pm;
      pm.x = gt * vv.x + (1.f - gt) * hh.x;
      pm.y = gt * vv.y + (1.f - gt) * hh.y;
      pm.z = gt * vv.z + (1.f - gt) * hh.z;
      pm.w = gt * vv.w + (1.f - gt) * hh.w;
      pm4[i] = pm;
      union { _Float16 h[4]; uint2 u; } pk;
      pk.h[0] = (_Float16)pm.x; pk.h[1] = (_Float16)pm.y;
      pk.h[2] = (_Float16)pm.z; pk.h[3] = (_Float16)pm.w;
      *(uint2*)(&s_pm[n * PSTRIDE + d4 * 4]) = pk.u;
    }
  }
  __syncthreads();

  // ---- Phase 5: scorer layer 1  [32,256] @ [256,64] via WMMA -> tanh -> s_att
  {
    const int nt = wave;           // 0..3
    v8f acc[2] = {};
    const uint32_t* wsa = wfrag + 32768;
    const uint32_t* r0 = (const uint32_t*)(s_pm + ml * PSTRIDE);
    const uint32_t* r1 = (const uint32_t*)(s_pm + (16 + ml) * PSTRIDE);
    for (int kt = 0; kt < 8; ++kt) {
      Frag a0, a1, bb;
      int kd = (kt * 32 + s * 8) >> 1;
#pragma unroll
      for (int j = 0; j < 4; ++j) {
        a0.u[j] = r0[kd + j];  a0.u[4 + j] = r0[kd + 8 + j];
        a1.u[j] = r1[kd + j];  a1.u[4 + j] = r1[kd + 8 + j];
      }
      const uint32_t* bp = wsa + (((nt * 8 + kt) * 32 + lane) << 3);
#pragma unroll
      for (int j = 0; j < 8; ++j) bb.u[j] = bp[j];
      acc[0] = __builtin_amdgcn_wmma_f32_16x16x32_f16(false, a0.h, false, bb.h, (short)0, acc[0], false, false);
      acc[1] = __builtin_amdgcn_wmma_f32_16x16x32_f16(false, a1.h, false, bb.h, (short)0, acc[1], false, false);
    }
    int col = nt * 16 + ml;
    float bias = ab1[col];
#pragma unroll
    for (int mt = 0; mt < 2; ++mt)
#pragma unroll
      for (int r = 0; r < 8; ++r) {
        int m = mt * 16 + r + 8 * s;
        s_att[m * ASTRIDE + col] = (_Float16)tanhf(acc[mt][r] + bias);
      }
  }
  __syncthreads();

  // ---- Phase 6a: scorer layer 2 (64-dot) + mask -> logits
  {
    int tok = tid >> 2, q = tid & 3;
    float p = 0.f;
    const _Float16* row = s_att + tok * ASTRIDE + q * 16;
#pragma unroll
    for (int i = 0; i < 16; ++i) p += (float)row[i] * aw2[q * 16 + i];
    p += __shfl_xor(p, 1, 32);
    p += __shfl_xor(p, 2, 32);
    if (q == 0) {
      float lg = p + ab2[0];
      bool mk = mask[(b * 32 + tok) * 512 + l] != 0;
      s_logits[tok] = mk ? lg : -__builtin_inff();
    }
  }
  __syncthreads();

  // ---- Phase 6b: masked softmax over the 32 messages (wave 0)
  if (wave == 0) {
    float lg = s_logits[lane];
    float mx = lg;
#pragma unroll
    for (int o = 16; o >= 1; o >>= 1) mx = fmaxf(mx, __shfl_xor(mx, o, 32));
    float e = (lg > -__builtin_inff()) ? __expf(lg - mx) : 0.f;
    float sm = e;
#pragma unroll
    for (int o = 16; o >= 1; o >>= 1) sm += __shfl_xor(sm, o, 32);
    float w = (sm > 0.f) ? e / sm : 0.f;   // fully-masked rows -> 0 (nan_to_num)
    s_attnw[lane] = w;
    out[OUT_ATTN + (size_t)b * 16384 + l * 32 + lane] = w;
  }
  __syncthreads();

  // ---- Phase 7: fused[d] = sum_n attn[n] * per_msg[n][d]
  {
    int d0 = tid * 2;
    float a0 = 0.f, a1 = 0.f;
#pragma unroll
    for (int n = 0; n < 32; ++n) {
      float w = s_attnw[n];
      a0 += w * (float)s_pm[n * PSTRIDE + d0];
      a1 += w * (float)s_pm[n * PSTRIDE + d0 + 1];
    }
    float2 f; f.x = a0; f.y = a1;
    *(float2*)(out + (size_t)g * 256 + d0) = f;
  }
}

extern "C" void kernel_launch(void* const* d_in, const int* in_sizes, int n_in,
                              void* d_out, int out_size, void* d_ws, size_t ws_size,
                              hipStream_t stream) {
  const float* vert  = (const float*)d_in[0];
  const float* horiz = (const float*)d_in[1];
  const unsigned char* mask = (const unsigned char*)d_in[2];
  const float* gw1 = (const float*)d_in[3];
  const float* gb1 = (const float*)d_in[4];
  const float* gw2 = (const float*)d_in[5];
  const float* gb2 = (const float*)d_in[6];
  const float* aw1 = (const float*)d_in[7];
  const float* ab1 = (const float*)d_in[8];
  const float* aw2 = (const float*)d_in[9];
  const float* ab2 = (const float*)d_in[10];
  uint32_t* ws = (uint32_t*)d_ws;           // needs 163,840 bytes
  float* out = (float*)d_out;

  prep_weights<<<160, 256, 0, stream>>>(gw1, aw1, ws);
  fusion_kernel<<<512 * 8, 128, 0, stream>>>(vert, horiz, mask, gb1, gw2, gb2,
                                             ab1, aw2, ab2, ws, out);
}
